// EGNNLayer_44925357916700
// MI455X (gfx1250) — compile-verified, matrix operand order checked by reading, not run on hardware
//
#include <hip/hip_runtime.h>
#include <hip/hip_bf16.h>

#define HD 128
#define NN 256
#define BB 8
#define BN (BB * NN)

typedef __attribute__((ext_vector_type(16))) _Float16 v16h;
typedef __attribute__((ext_vector_type(8)))  float    v8f;

union FragA { v16h v; _Float16 h[16]; int4 q[2]; };
union FragC { v8f  v; float f[8]; };

__device__ __forceinline__ float clamp50(float x) { return fminf(fmaxf(x, -50.f), 50.f); }
__device__ __forceinline__ float clamp20(float x) { return fminf(fmaxf(x, -20.f), 20.f); }
__device__ __forceinline__ float silu(float x) { return x / (1.f + __expf(-x)); }

// B fragment (K x 16 slab of W^T, i.e. B[k,n] = W[n,k]) from fp32 weights.
// Mirrors the 16-bit A layout: lane n = l&15; lanes>=16 take K-offset +8;
// frag.h[0..7] = K kb..kb+7, frag.h[8..15] = K kb+16..kb+23.
__device__ __forceinline__ void load_bfrag_f32(FragA& f, const float* __restrict__ W,
                                               int ldw, int n, int kb) {
  const float* r = W + n * ldw + kb;
#pragma unroll
  for (int e = 0; e < 8; ++e) {
    f.h[e]     = (_Float16)r[e];
    f.h[8 + e] = (_Float16)r[16 + e];
  }
}

// A fragment from a row-major f16 LDS tile: lane's row = l&15, K base kb (16B aligned).
__device__ __forceinline__ void load_afrag(FragA& f, const _Float16* As, int lda,
                                           int arow, int kb) {
  f.q[0] = *(const int4*)(As + arow * lda + kb);
  f.q[1] = *(const int4*)(As + arow * lda + kb + 16);
}

__device__ __forceinline__ v8f wmma16(const FragA& a, const FragA& b, const v8f& c) {
  return __builtin_amdgcn_wmma_f32_16x16x32_f16(false, a.v, false, b.v, (short)0, c,
                                                false, false);
}

// ---------------------------------------------------------------------------
// Kernel 1: Q[b,i] = W1a @ clamp(h_i), P[b,j] = W1b @ clamp(h_j); coords_out.
// grid = BN/16 workgroups, 256 threads (8 waves). Each WG = 16 node rows.
// ---------------------------------------------------------------------------
__global__ void k_precompute(const float* __restrict__ h, const float* __restrict__ w1e,
                             const float* __restrict__ coords, const float* __restrict__ mask,
                             float* __restrict__ Qc, float* __restrict__ Pc,
                             float* __restrict__ coords_out) {
  __shared__ __align__(16) _Float16 As[16 * HD];
  const int t = threadIdx.x;
  const int lane = t & 31, wv = t >> 5;
  const int nloc = lane & 15, khalf = (lane >> 4) << 3, arow = lane & 15;
  const int rowBase = blockIdx.x * 16;

  // clamp(h) -> f16 A tile
#pragma unroll
  for (int rep = 0; rep < 8; ++rep) {
    int idx = rep * 256 + t;
    int row = idx >> 7, col = idx & 127;
    As[row * HD + col] = (_Float16)clamp50(h[(rowBase + row) * HD + col]);
  }
  // coords_out = clamp(coords,20) * mask  (16 nodes * 3)
  if (t < 48) {
    int node = rowBase + t / 3, c3 = t % 3;
    coords_out[node * 3 + c3] = clamp20(coords[node * 3 + c3]) * mask[node];
  }

  // B fragments: W1a = w1e[:,0:128], W1b = w1e[:,128:256] (row stride 257)
  FragA bq[4], bp[4];
#pragma unroll
  for (int kc = 0; kc < 4; ++kc) {
    load_bfrag_f32(bq[kc], w1e, 257, wv * 16 + nloc, kc * 32 + khalf);
    load_bfrag_f32(bp[kc], w1e + 128, 257, wv * 16 + nloc, kc * 32 + khalf);
  }
  __syncthreads();

  FragC cq, cp;
#pragma unroll
  for (int i = 0; i < 8; ++i) { cq.f[i] = 0.f; cp.f[i] = 0.f; }
  FragA a;
#pragma unroll
  for (int kc = 0; kc < 4; ++kc) {
    load_afrag(a, As, HD, arow, kc * 32 + khalf);
    cq.v = wmma16(a, bq[kc], cq.v);
    cp.v = wmma16(a, bp[kc], cp.v);
  }
#pragma unroll
  for (int r = 0; r < 8; ++r) {
    int row = r + khalf;
    Qc[(rowBase + row) * HD + wv * 16 + nloc] = cq.f[r];
    Pc[(rowBase + row) * HD + wv * 16 + nloc] = cp.f[r];
  }
}

// ---------------------------------------------------------------------------
// Kernel 2: per (b,i) aggregate messages over all j.
// grid = BN workgroups, 256 threads. Wave w owns output features [16w,16w+16).
// Double-buffered activation tile: one barrier per j-tile; build(jt+1)
// overlaps (global Pc loads + SiLU VALU) with the WMMAs on tile jt.
// ---------------------------------------------------------------------------
__global__ void k_edges(const float* __restrict__ coords, const float* __restrict__ mask,
                        const float* __restrict__ w1e, const float* __restrict__ b1e,
                        const float* __restrict__ w2e, const float* __restrict__ b2e,
                        const float* __restrict__ Qc, const float* __restrict__ Pc,
                        float* __restrict__ magg) {
  __shared__ __align__(16) _Float16 As[2][16 * HD];
  __shared__ __align__(16) float mrow[NN];
  __shared__ float preQ[HD];
  __shared__ float w1c[HD];
  __shared__ float d2a[NN];
  __shared__ float sred[256];
  __shared__ float cnt;

  const int t = threadIdx.x;
  const int bi = blockIdx.x;           // = b*NN + i
  const int b = bi >> 8;
  const float mi = mask[bi];

  if (mi == 0.f) {                     // whole-node masked: m_agg = 0 (uniform branch)
    if (t < HD) magg[bi * HD + t] = 0.f;
    return;
  }

  const int lane = t & 31, wv = t >> 5;
  const int nloc = lane & 15, khalf = (lane >> 4) << 3, arow = lane & 15;

  const float cix = clamp20(coords[bi * 3 + 0]);
  const float ciy = clamp20(coords[bi * 3 + 1]);
  const float ciz = clamp20(coords[bi * 3 + 2]);

  if (t == 0) cnt = 0.f;
  if (t < HD) {
    preQ[t] = Qc[bi * HD + t] + b1e[t];
    w1c[t] = w1e[t * 257 + 256];       // dist2 column of W1e
  }
  float mj;
  {
    int j = t;
    mj = mask[b * NN + j];
    mrow[j] = mj;
    float dx = cix - clamp20(coords[(b * NN + j) * 3 + 0]);
    float dy = ciy - clamp20(coords[(b * NN + j) * 3 + 1]);
    float dz = ciz - clamp20(coords[(b * NN + j) * 3 + 2]);
    d2a[j] = fminf(dx * dx + dy * dy + dz * dz, 50.f);  // clip(.,400) then clamp 50
  }

  // W2e B fragments, resident in VGPRs for the whole loop
  FragA bw[4];
#pragma unroll
  for (int kc = 0; kc < 4; ++kc)
    load_bfrag_f32(bw[kc], w2e, HD, wv * 16 + nloc, kc * 32 + khalf);
  const float b2 = b2e[wv * 16 + nloc];

  __syncthreads();
  atomicAdd(&cnt, mj);                 // neighbor count (exact 0/1 sum)

  const int fcol = t & 127;
  const int jbase = t >> 7;
  const float preQr = preQ[fcol];      // hoisted uniform-per-thread LDS reads
  const float w1cr = w1c[fcol];
  const float* __restrict__ Pcb = Pc + (size_t)(b * NN) * HD + fcol;

  auto build = [&](int jt, _Float16* __restrict__ dst) {
#pragma unroll
    for (int rep = 0; rep < 8; ++rep) {
      int jl = jbase + rep * 2;
      int j = jt * 16 + jl;
      float pre = clamp50(preQr + Pcb[j * HD] + w1cr * d2a[j]);
      dst[jl * HD + fcol] = (_Float16)silu(pre);
    }
  };

  build(0, As[0]);
  __syncthreads();

  float acc = 0.f;
  for (int jt = 0; jt < 16; ++jt) {
    const _Float16* cur = As[jt & 1];
    if (jt < 15) build(jt + 1, As[(jt + 1) & 1]);   // overlaps with WMMAs below

    FragC d;
#pragma unroll
    for (int i = 0; i < 8; ++i) d.f[i] = 0.f;
    FragA a;
#pragma unroll
    for (int kc = 0; kc < 4; ++kc) {
      load_afrag(a, cur, HD, arow, kc * 32 + khalf);
      d.v = wmma16(a, bw[kc], d.v);
    }
    // clamp + edge-mask + accumulate over j (mask read as two b128 loads)
    const float4 m0 = *(const float4*)&mrow[jt * 16 + khalf];
    const float4 m1 = *(const float4*)&mrow[jt * 16 + khalf + 4];
    acc += clamp50(d.f[0] + b2) * m0.x + clamp50(d.f[1] + b2) * m0.y +
           clamp50(d.f[2] + b2) * m0.z + clamp50(d.f[3] + b2) * m0.w +
           clamp50(d.f[4] + b2) * m1.x + clamp50(d.f[5] + b2) * m1.y +
           clamp50(d.f[6] + b2) * m1.z + clamp50(d.f[7] + b2) * m1.w;
    __syncthreads();
  }

  // combine the two j-halves (lane l and l+16 hold the same feature column)
  sred[t] = acc;
  __syncthreads();
  if (lane < 16) {
    float tot = sred[wv * 32 + lane] + sred[wv * 32 + lane + 16];
    float denom = fmaxf(cnt, 1.f);
    magg[bi * HD + wv * 16 + lane] = clamp50(tot / denom);
  }
}

// ---------------------------------------------------------------------------
// Kernel 3: node MLP on concat(clamp(h), m_agg): 256->128 SiLU 128->128, masked.
// grid = BN/16 workgroups, 256 threads.
// ---------------------------------------------------------------------------
__global__ void k_nodes(const float* __restrict__ h, const float* __restrict__ mask,
                        const float* __restrict__ magg,
                        const float* __restrict__ w1n, const float* __restrict__ b1n,
                        const float* __restrict__ w2n, const float* __restrict__ b2n,
                        float* __restrict__ h_out) {
  __shared__ __align__(16) _Float16 A1[16 * 256];
  __shared__ __align__(16) _Float16 A2[16 * HD];
  const int t = threadIdx.x;
  const int lane = t & 31, wv = t >> 5;
  const int nloc = lane & 15, khalf = (lane >> 4) << 3, arow = lane & 15;
  const int rowBase = blockIdx.x * 16;

  // A1 tile: cols 0..127 = clamp(h,50), cols 128..255 = m_agg (already clamped)
#pragma unroll
  for (int row = 0; row < 16; ++row) {
    int col = t;
    float x = (col < HD) ? clamp50(h[(rowBase + row) * HD + col])
                         : magg[(rowBase + row) * HD + (col - HD)];
    A1[row * 256 + col] = (_Float16)x;
  }

  // layer-1 B fragments (K = 256)
  FragA b1f[8];
#pragma unroll
  for (int kc = 0; kc < 8; ++kc)
    load_bfrag_f32(b1f[kc], w1n, 256, wv * 16 + nloc, kc * 32 + khalf);
  const float bb1 = b1n[wv * 16 + nloc];
  __syncthreads();

  FragC d1;
#pragma unroll
  for (int i = 0; i < 8; ++i) d1.f[i] = 0.f;
  FragA a;
#pragma unroll
  for (int kc = 0; kc < 8; ++kc) {
    load_afrag(a, A1, 256, arow, kc * 32 + khalf);
    d1.v = wmma16(a, b1f[kc], d1.v);
  }
  // clamp + SiLU -> A2 relay tile
#pragma unroll
  for (int r = 0; r < 8; ++r) {
    int row = r + khalf;
    float x = clamp50(d1.f[r] + bb1);
    A2[row * HD + wv * 16 + nloc] = (_Float16)silu(x);
  }

  // layer-2 B fragments (K = 128)
  FragA b2f[4];
#pragma unroll
  for (int kc = 0; kc < 4; ++kc)
    load_bfrag_f32(b2f[kc], w2n, HD, wv * 16 + nloc, kc * 32 + khalf);
  const float bb2 = b2n[wv * 16 + nloc];
  __syncthreads();

  FragC d2;
#pragma unroll
  for (int i = 0; i < 8; ++i) d2.f[i] = 0.f;
#pragma unroll
  for (int kc = 0; kc < 4; ++kc) {
    load_afrag(a, A2, HD, arow, kc * 32 + khalf);
    d2.v = wmma16(a, b2f[kc], d2.v);
  }
#pragma unroll
  for (int r = 0; r < 8; ++r) {
    int row = r + khalf;
    int node = rowBase + row;
    h_out[node * HD + wv * 16 + nloc] = clamp50(d2.f[r] + bb2) * mask[node];
  }
}

// ---------------------------------------------------------------------------
extern "C" void kernel_launch(void* const* d_in, const int* in_sizes, int n_in,
                              void* d_out, int out_size, void* d_ws, size_t ws_size,
                              hipStream_t stream) {
  (void)in_sizes; (void)n_in; (void)out_size; (void)ws_size;
  const float* h      = (const float*)d_in[0];
  const float* coords = (const float*)d_in[1];
  const float* mask   = (const float*)d_in[2];
  const float* w1e    = (const float*)d_in[3];
  const float* b1e    = (const float*)d_in[4];
  const float* w2e    = (const float*)d_in[5];
  const float* b2e    = (const float*)d_in[6];
  const float* w1n    = (const float*)d_in[7];
  const float* b1n    = (const float*)d_in[8];
  const float* w2n    = (const float*)d_in[9];
  const float* b2n    = (const float*)d_in[10];

  float* h_out      = (float*)d_out;              // [B,N,128]
  float* coords_out = h_out + BN * HD;            // [B,N,3]

  float* Qc = (float*)d_ws;                       // [BN,128]
  float* Pc = Qc + BN * HD;                       // [BN,128]
  float* mg = Pc + BN * HD;                       // [BN,128]

  k_precompute<<<BN / 16, 256, 0, stream>>>(h, w1e, coords, mask, Qc, Pc, coords_out);
  k_edges<<<BN, 256, 0, stream>>>(coords, mask, w1e, b1e, w2e, b2e, Qc, Pc, mg);
  k_nodes<<<BN / 16, 256, 0, stream>>>(h, mask, mg, w1n, b1n, w2n, b2n, h_out);
}
